// Attention_Head_40441412059265
// MI455X (gfx1250) — matrix operand order repaired and checked by reading, not compiled, not run on hardware
//
#include <hip/hip_runtime.h>

// ---------------------------------------------------------------------------
// Attention head for MI455X (gfx1250): bf16 WMMA for QKV projection,
// QK^T scores, and P*V, with flash-attention online softmax.
// Shapes: B=4, S=4096, D_IN=512, D_OUT=64. scale = 1/sqrt(64) = 0.125.
// ---------------------------------------------------------------------------

typedef __attribute__((ext_vector_type(16))) __bf16 v16bf;
typedef __attribute__((ext_vector_type(8)))  __bf16 v8bf;
typedef __attribute__((ext_vector_type(8)))  float  v8f;

#define BATCH 4
#define SLEN  4096
#define DIN   512
#define DOUT  64
#define NTOK  (BATCH * SLEN)          // 16384
#define SM_SCALE 0.125f               // 1/sqrt(64)

// Native conversions: let the compiler emit packed v_cvt bf16 ops.
static __device__ __forceinline__ __bf16 f32_to_bf16(float f) {
    return (__bf16)f;
}
static __device__ __forceinline__ v8bf cvt8_bf16(v8f x) {
    return __builtin_convertvector(x, v8bf);
}

static __device__ __forceinline__ v8f wmma_bf16(v16bf a, v16bf b, v8f c) {
    // D = A(16x32 bf16) * B(32x16 bf16) + C(16x16 f32)
    return __builtin_amdgcn_wmma_f32_16x16x32_bf16(
        /*neg_a=*/false, a, /*neg_b=*/false, b,
        /*c_mod=*/(short)0, c, /*reuse_a=*/false, /*reuse_b=*/false);
}

static __device__ __forceinline__ v16bf cat8(v8bf lo, v8bf hi) {
    return __builtin_shufflevector(lo, hi, 0,1,2,3,4,5,6,7,8,9,10,11,12,13,14,15);
}

// ---------------------------------------------------------------------------
// Kernel 1: transpose + convert weights to bf16.  Wt[m][n][k] = W_m[k][n].
// ---------------------------------------------------------------------------
__global__ void ah_wconv(const float* __restrict__ Wq,
                         const float* __restrict__ Wk,
                         const float* __restrict__ Wv,
                         __bf16* __restrict__ Wt) {
    int idx = blockIdx.x * 256 + threadIdx.x;     // 0 .. 3*64*512-1
    if (idx >= 3 * DOUT * DIN) return;
    int m = idx / (DOUT * DIN);
    int r = idx % (DOUT * DIN);
    int n = r / DIN;
    int k = r % DIN;
    const float* W = (m == 0) ? Wq : (m == 1) ? Wk : Wv;
    Wt[idx] = f32_to_bf16(W[(size_t)k * DOUT + n]);
}

// ---------------------------------------------------------------------------
// Kernel 2: QKV projection.  One wave computes a 16x64 output row-block with
// 16 WMMA k-steps (k=32 each) over D_IN=512.
//   mat 0: Q[t][64]   (row-major bf16)
//   mat 1: K[t][64]   (row-major bf16)
//   mat 2: Vt[b][64][S] (transposed bf16)
// ---------------------------------------------------------------------------
__global__ __launch_bounds__(32)
void ah_qkv_proj(const float* __restrict__ Xq,
                 const float* __restrict__ Xk,
                 const float* __restrict__ Xv,
                 const __bf16* __restrict__ Wt,
                 __bf16* __restrict__ Q,
                 __bf16* __restrict__ K,
                 __bf16* __restrict__ Vt) {
    const int tile = blockIdx.x;            // 0 .. 3*1024-1
    const int mat  = tile / (NTOK / 16);    // 0=Q,1=K,2=V
    const int rt   = tile % (NTOK / 16);    // 16-row tile over 16384 tokens
    const int lane = threadIdx.x;
    const int l16    = lane & 15;
    const int base8  = (lane >= 16) ? 8 : 0;
    const int base16 = (lane >= 16) ? 16 : 0;

    const float* X = (mat == 0) ? Xq : (mat == 1) ? Xk : Xv;
    const __bf16* W = Wt + (size_t)mat * DOUT * DIN;

    const int row = rt * 16 + l16;          // A-matrix row owned by this lane

    v8f acc[4] = {};                        // 4 column tiles (16x16 each)

    #pragma unroll 4
    for (int kb = 0; kb < DIN; kb += 32) {
        // A fragment (16x32): lane holds row `row`, K = base16 .. base16+15
        // within this k-step.
        const float* xp = X + (size_t)row * DIN + kb;
        v8f x0 = *(const v8f*)(xp + base16);
        v8f x1 = *(const v8f*)(xp + base16 + 8);
        v16bf a = cat8(cvt8_bf16(x0), cvt8_bf16(x1));
        #pragma unroll
        for (int t = 0; t < 4; t++) {
            // B fragment (32x16): col n = t*16+l16, K = base16+e  (contiguous in Wt)
            v16bf b = *(const v16bf*)(W + (size_t)(t * 16 + l16) * DIN + kb + base16);
            acc[t] = wmma_bf16(a, b, acc[t]);
        }
    }

    // C layout: lane owns col N = l16, rows M = base8+e.
    const int rowBase = rt * 16 + base8;
    if (mat < 2) {
        __bf16* O = (mat == 0) ? Q : K;
        #pragma unroll
        for (int t = 0; t < 4; t++) {
            v8bf o = cvt8_bf16(acc[t]);
            #pragma unroll
            for (int e = 0; e < 8; e++)
                O[(size_t)(rowBase + e) * DOUT + t * 16 + l16] = o[e];
        }
    } else {
        // Vt[b][dim][s]; 8 consecutive tokens stay inside one batch (4096%16==0)
        // -> one contiguous 16-byte store per tile.
        const int b  = rowBase >> 12;       // /4096
        const int s0 = rowBase & (SLEN - 1);
        #pragma unroll
        for (int t = 0; t < 4; t++) {
            const int dim = t * 16 + l16;
            *(v8bf*)(Vt + ((size_t)b * DOUT + dim) * SLEN + s0) = cvt8_bf16(acc[t]);
        }
    }
}

// ---------------------------------------------------------------------------
// Kernel 3: causal flash attention.  One wave owns 16 query rows, loops over
// 32-key blocks up to the diagonal.  Per block: 4 score WMMAs, LDS-based
// online softmax (also re-lays-out P from C-layout to A-layout), 4 PV WMMAs.
// ---------------------------------------------------------------------------
__global__ __launch_bounds__(32)
void ah_flash_attn(const __bf16* __restrict__ Q,
                   const __bf16* __restrict__ K,
                   const __bf16* __restrict__ Vt,
                   float* __restrict__ Out) {
    __shared__ __align__(32) float  sS[16 * 32];   // scores tile (f32)
    __shared__ __align__(32) __bf16 sP[16 * 32];   // probs tile (bf16, A-layout src)
    __shared__ float sCorr[16];                    // per-row rescale factor
    __shared__ float sInvL[16];                    // per-row 1/l at the end

    const int tileId = blockIdx.x;                 // 0..1023
    const int b      = tileId >> 8;                // /256
    const int qbase  = (tileId & 255) * 16;
    const int lane   = threadIdx.x;
    const int l16    = lane & 15;
    const int base8  = (lane >= 16) ? 8 : 0;
    const int base16 = (lane >= 16) ? 16 : 0;

    // --- load Q fragments for kb=0 and kb=32 (kept in registers) ---
    const __bf16* qp = Q + ((size_t)b * SLEN + qbase + l16) * DOUT;
    v16bf aQ0 = cat8(*(const v8bf*)(qp + base16),      *(const v8bf*)(qp + 8 + base16));
    v16bf aQ1 = cat8(*(const v8bf*)(qp + 32 + base16), *(const v8bf*)(qp + 40 + base16));

    v8f acc[4] = {};
    float m_r = -1e30f, l_r = 0.0f;                // valid in lanes 0..15

    const int kend = qbase + 15;
    for (int kt = 0; kt <= kend; kt += 32) {
        // --- scores: S = Q * K^T, two 16x16 tiles (keys kt..kt+15, kt+16..kt+31)
        v8f c0 = {}, c1 = {};
        {
            const __bf16* kp0 = K + ((size_t)b * SLEN + kt + l16) * DOUT;
            const __bf16* kp1 = K + ((size_t)b * SLEN + kt + 16 + l16) * DOUT;
            v16bf b00 = *(const v16bf*)(kp0 + base16);       // dims 0..31
            v16bf b01 = *(const v16bf*)(kp0 + 32 + base16);  // dims 32..63
            v16bf b10 = *(const v16bf*)(kp1 + base16);
            v16bf b11 = *(const v16bf*)(kp1 + 32 + base16);
            c0 = wmma_bf16(aQ0, b00, c0);
            c0 = wmma_bf16(aQ1, b01, c0);
            c1 = wmma_bf16(aQ0, b10, c1);
            c1 = wmma_bf16(aQ1, b11, c1);
        }

        // --- scores -> LDS (C layout: lane owns col l16, rows base8+e)
        #pragma unroll
        for (int e = 0; e < 8; e++) {
            sS[(base8 + e) * 32 + l16]      = c0[e];
            sS[(base8 + e) * 32 + 16 + l16] = c1[e];
        }
        __syncthreads();

        // --- online softmax: lanes 0..15 each own one query row
        if (lane < 16) {
            const int q = qbase + lane;
            float mx = -1e30f;
            #pragma unroll
            for (int c = 0; c < 32; c++) {
                float s = (kt + c > q) ? -1e30f : sS[lane * 32 + c] * SM_SCALE;
                mx = fmaxf(mx, s);
            }
            const float mnew = fmaxf(m_r, mx);
            const float corr = __expf(m_r - mnew);   // 0 on first block
            float sum = 0.0f;
            #pragma unroll
            for (int c = 0; c < 32; c++) {
                float p = (kt + c > q) ? 0.0f
                                       : __expf(sS[lane * 32 + c] * SM_SCALE - mnew);
                sum += p;
                sP[lane * 32 + c] = f32_to_bf16(p);
            }
            l_r = l_r * corr + sum;
            m_r = mnew;
            sCorr[lane] = corr;
        }
        __syncthreads();

        // --- rescale accumulators by per-row correction
        #pragma unroll
        for (int e = 0; e < 8; e++) {
            const float corr = sCorr[base8 + e];
            #pragma unroll
            for (int t = 0; t < 4; t++) acc[t][e] *= corr;
        }

        // --- P (16x32) * V (32x64): A-fragment of P from LDS, B from Vt
        v16bf aP = cat8(*(const v8bf*)(&sP[l16 * 32 + base16]),
                        *(const v8bf*)(&sP[l16 * 32 + 8 + base16]));
        #pragma unroll
        for (int t = 0; t < 4; t++) {
            const __bf16* vp = Vt + ((size_t)b * DOUT + t * 16 + l16) * SLEN + kt + base16;
            v16bf bV = *(const v16bf*)vp;
            acc[t] = wmma_bf16(aP, bV, acc[t]);
        }
        // no extra barrier needed: next block's sS/sCorr writes are fenced by
        // the two barriers above relative to this block's reads.
    }

    if (lane < 16) sInvL[lane] = 1.0f / l_r;
    __syncthreads();

    #pragma unroll
    for (int t = 0; t < 4; t++) {
        const int dim = t * 16 + l16;
        #pragma unroll
        for (int e = 0; e < 8; e++) {
            const int r = base8 + e;
            Out[((size_t)b * SLEN + qbase + r) * DOUT + dim] = acc[t][e] * sInvL[r];
        }
    }
}

// ---------------------------------------------------------------------------
// Host-side launcher.
// d_in order: key_inputs, value_inputs, query_inputs, Wq, Wk, Wv
// workspace: Wt bf16[3*64*512] | Q bf16[16384*64] | K bf16[16384*64] | Vt bf16[4*64*4096]
// ---------------------------------------------------------------------------
extern "C" void kernel_launch(void* const* d_in, const int* in_sizes, int n_in,
                              void* d_out, int out_size, void* d_ws, size_t ws_size,
                              hipStream_t stream) {
    const float* Xk = (const float*)d_in[0];
    const float* Xv = (const float*)d_in[1];
    const float* Xq = (const float*)d_in[2];
    const float* Wq = (const float*)d_in[3];
    const float* Wk = (const float*)d_in[4];
    const float* Wv = (const float*)d_in[5];
    float* Out = (float*)d_out;

    char* ws = (char*)d_ws;
    const size_t WT_BYTES  = (size_t)3 * DOUT * DIN * 2;      // 196608
    const size_t MAT_BYTES = (size_t)NTOK * DOUT * 2;         // 2 MiB each
    __bf16* Wt = (__bf16*)ws;
    __bf16* Qb = (__bf16*)(ws + WT_BYTES);
    __bf16* Kb = (__bf16*)(ws + WT_BYTES + MAT_BYTES);
    __bf16* Vt = (__bf16*)(ws + WT_BYTES + 2 * MAT_BYTES);

    ah_wconv<<<(3 * DOUT * DIN) / 256, 256, 0, stream>>>(Wq, Wk, Wv, Wt);
    ah_qkv_proj<<<3 * (NTOK / 16), 32, 0, stream>>>(Xq, Xk, Xv, Wt, Qb, Kb, Vt);
    ah_flash_attn<<<BATCH * (SLEN / 16), 32, 0, stream>>>(Qb, Kb, Vt, Out);
}
